// NeuralTensorNetwork_23991687315870
// MI455X (gfx1250) — matrix operand order repaired and checked by reading, not compiled, not run on hardware
//
#include <hip/hip_runtime.h>
#include <math.h>

// MI455X / gfx1250 Neural Tensor Network.
//
// bil[k,b] = e1[b]·W[k]·e2[b] (34.4 GFLOP fp32) as split-precision bf16 WMMA:
// x = hi + lo (bf16 pair), acc += Ah*Bh + Ah*Bl + Al*Bh in fp32 accumulators
// (~fp32 accuracy at bf16 tensor-core rates).  Each wave owns one W[k]
// permanently in VGPRs and streams 16-row batch tiles, no inter-wave sync.
//
// A prep kernel builds, once per element (instead of once per k-wave):
//   * e1 split to bf16 hi/lo in the exact WMMA A-operand swizzle (32 B/lane)
//   * e2 permuted into the WMMA C/D lane layout for the fp32 row-dot epilogue
// so the hot loop is pure b128 loads + 24 WMMAs + a short fp32 epilogue.
// Falls back to inline gather/convert when ws_size < nTiles*8 KB.

typedef __attribute__((ext_vector_type(16))) __bf16 v16bf;
typedef __attribute__((ext_vector_type(8)))  float  v8f;

static __device__ __forceinline__ void bsplit(v16bf& hi, v16bf& lo, int e, float x) {
  __bf16 h = (__bf16)x;
  hi[e] = h;
  lo[e] = (__bf16)(x - (float)h);
}

// WMMA A operand (16x32, bf16) hi/lo from a row-major fp32 matrix.
// ISA layout: lane&15 = M; lanes<16 hold K 0-7 & 16-23, lanes>=16 hold 8-15 & 24-31.
static __device__ __forceinline__ void buildA(const float* __restrict__ base, int row0,
                                              int ld, int colbase, int lane,
                                              v16bf& hi, v16bf& lo) {
  const float* p = base + (size_t)(row0 + (lane & 15)) * ld + colbase + ((lane < 16) ? 0 : 8);
#pragma unroll
  for (int e = 0; e < 8; ++e) bsplit(hi, lo, e, p[e]);
#pragma unroll
  for (int e = 0; e < 8; ++e) bsplit(hi, lo, 8 + e, p[16 + e]);
}

// Pre-split e1 A-operand cache: [tile][kc][lane] -> 16 bf16 (32 B).
static __device__ __forceinline__ size_t wsOffA(int tile, int kc, int lane) {
  return ((size_t)(tile * 2 + kc) * 32 + (size_t)lane) * 16;
}
// Pre-permuted e2 epilogue cache: [tile][t][lane] -> 8 fp32 (32 B).
static __device__ __forceinline__ size_t wsOffE2(int tile, int t, int lane) {
  return ((size_t)(tile * 4 + t) * 32 + (size_t)lane) * 8;
}

static __device__ __forceinline__ void loadA(const __bf16* __restrict__ wsHi,
                                             const __bf16* __restrict__ wsLo,
                                             int tile, int kc, int lane,
                                             v16bf& hi, v16bf& lo) {
  const size_t off = wsOffA(tile, kc, lane);
  hi = *(const v16bf*)(wsHi + off);
  lo = *(const v16bf*)(wsLo + off);
}

// WMMA B operand (32x16, bf16) hi/lo from a row-major fp32 [K x N] matrix.
// ISA layout: lanes 0-15 hold K=i0+0-7 & 16-23, lanes 16-31 hold K=i0+8-15 & 24-31; N = lane&15.
static __device__ __forceinline__ void buildB(const float* __restrict__ M, int ld,
                                              int i0, int j, int lane,
                                              v16bf& hi, v16bf& lo) {
  const float* p = M + (size_t)(i0 + ((lane < 16) ? 0 : 8)) * ld + j;
#pragma unroll
  for (int e = 0; e < 16; ++e) bsplit(hi, lo, e, p[(size_t)(e < 8 ? e : e + 8) * ld]);
}

// Split-precision product: acc += (Ah+Al)*(Bh+Bl), dropping only Al*Bl (~1e-5 rel).
static __device__ __forceinline__ v8f wmma3(v16bf ah, v16bf al, v16bf bh, v16bf bl, v8f acc) {
  acc = __builtin_amdgcn_wmma_f32_16x16x32_bf16(false, al, false, bh, (short)0, acc, false, false);
  acc = __builtin_amdgcn_wmma_f32_16x16x32_bf16(false, ah, false, bl, (short)0, acc, false, false);
  acc = __builtin_amdgcn_wmma_f32_16x16x32_bf16(false, ah, false, bh, (short)0, acc, false, false);
  return acc;
}

// ---------------------------------------------------------------------------
// Kernel 0 (prep): e1 -> bf16 hi/lo in A-operand swizzle; e2 -> fp32 in the
// C/D lane layout used by the epilogue.  All stores are 32 B/lane, coalesced.
// ---------------------------------------------------------------------------
__global__ __launch_bounds__(128, 1)
void ntn_prep_kernel(const float* __restrict__ e1, const float* __restrict__ e2,
                     __bf16* __restrict__ wsHi, __bf16* __restrict__ wsLo,
                     float* __restrict__ e2perm, int nTiles) {
  const int lane  = threadIdx.x & 31;
  const int gwave = blockIdx.x * (blockDim.x >> 5) + (threadIdx.x >> 5);
  const int nWav  = gridDim.x * (blockDim.x >> 5);
  const int c     = lane & 15;
  const int hoff  = (lane < 16) ? 0 : 8;

  for (int t = gwave; t < nTiles; t += nWav) {
#pragma unroll
    for (int kc = 0; kc < 2; ++kc) {
      v16bf hi, lo;
      buildA(e1, t * 16, 64, kc * 32, lane, hi, lo);
      const size_t off = wsOffA(t, kc, lane);
      *(v16bf*)(wsHi + off) = hi;
      *(v16bf*)(wsLo + off) = lo;
    }
#pragma unroll
    for (int nt = 0; nt < 4; ++nt) {           // e2 gathered once, not 16x
      const float* e2p = e2 + (size_t)(t * 16 + hoff) * 64 + nt * 16 + c;
      float buf[8];
#pragma unroll
      for (int v = 0; v < 8; ++v) buf[v] = e2p[(size_t)v * 64];
      float* dst = e2perm + wsOffE2(t, nt, lane);
      *(float4*)dst       = make_float4(buf[0], buf[1], buf[2], buf[3]);
      *(float4*)(dst + 4) = make_float4(buf[4], buf[5], buf[6], buf[7]);
    }
  }
}

// ---------------------------------------------------------------------------
// Kernel 1: ff = concat(e1,e2) @ V  ->  d_out   (memory-bound, WMMA-driven)
// ---------------------------------------------------------------------------
template <bool USE_WS>
__global__ __launch_bounds__(128, 1)
void ntn_ff_kernel(const float* __restrict__ e1, const float* __restrict__ e2,
                   const float* __restrict__ V, float* __restrict__ out,
                   const __bf16* __restrict__ wsHi, const __bf16* __restrict__ wsLo,
                   int nTiles) {
  const int lane  = threadIdx.x & 31;
  const int gwave = blockIdx.x * (blockDim.x >> 5) + (threadIdx.x >> 5);
  const int nWav  = gridDim.x * (blockDim.x >> 5);

  // Preload V (128x16) as 4 K-chunks of bf16 hi/lo B operands (held in VGPRs).
  v16bf vh[4], vl[4];
#pragma unroll
  for (int kc = 0; kc < 4; ++kc)
    buildB(V, 16, kc * 32, lane & 15, lane, vh[kc], vl[kc]);

  for (int t = gwave; t < nTiles; t += nWav) {
    const int b0 = t * 16;
    v8f acc = {};
#pragma unroll
    for (int kc = 0; kc < 4; ++kc) {           // K = 128 over concat(e1,e2)
      v16bf ah, al;
      if (kc < 2) {
        if (USE_WS) loadA(wsHi, wsLo, t, kc, lane, ah, al);
        else        buildA(e1, b0, 64, kc * 32, lane, ah, al);
      } else {
        buildA(e2, b0, 64, (kc & 1) * 32, lane, ah, al);
      }
      acc = wmma3(ah, al, vh[kc], vl[kc], acc);
    }
    // C/D layout: vgpr v -> row b0 + v + 8*(lane>=16), col = lane&15.
    const int rowoff = b0 + ((lane < 16) ? 0 : 8);
    const int c = lane & 15;
#pragma unroll
    for (int v = 0; v < 8; ++v)
      out[(size_t)(rowoff + v) * 16 + c] = acc[v];
  }
}

// ---------------------------------------------------------------------------
// Kernel 2: bilinear term via WMMA, fused permutation + bias + tanh.
// ---------------------------------------------------------------------------
template <bool USE_WS>
__global__ __launch_bounds__(128, 1)
void ntn_bil_kernel(const float* __restrict__ e1, const float* __restrict__ e2,
                    const float* __restrict__ W, const float* __restrict__ bias,
                    float* __restrict__ out,
                    const __bf16* __restrict__ wsHi, const __bf16* __restrict__ wsLo,
                    const float* __restrict__ e2perm, int nTiles) {
  __shared__ float xpose[4][16][16];            // per-wave 16x16 transpose (64B rows)
  const int lane = threadIdx.x & 31;
  const int wv   = threadIdx.x >> 5;
  const int gwave = blockIdx.x * 4 + wv;
  const int k     = gwave & 15;                 // each wave owns one W-slice
  const int tile0 = gwave >> 4;
  const int tileStride = (gridDim.x * 4) >> 4;

  // sum(bias) over 64 entries (added as a scalar in the reference).
  float bs = bias[lane] + bias[lane + 32];
#pragma unroll
  for (int m = 16; m >= 1; m >>= 1) bs += __shfl_xor(bs, m, 32);

  // Preload W[k] (64x64) as 4 N-tiles x 2 K-chunks of bf16 hi/lo (128 VGPRs).
  v16bf wh[4][2], wl[4][2];
  const float* Wk = W + (size_t)k * 64 * 64;
#pragma unroll
  for (int t = 0; t < 4; ++t)
#pragma unroll
    for (int kc = 0; kc < 2; ++kc)
      buildB(Wk, 64, kc * 32, t * 16 + (lane & 15), lane, wh[t][kc], wl[t][kc]);

  const int c    = lane & 15;
  const int hoff = (lane < 16) ? 0 : 8;

  for (int tile = tile0; tile < nTiles; tile += tileStride) {
    const int b0 = tile * 16;

    // Prefetch next tile's operands into cache.
    const int nt = tile + tileStride;
    if (nt < nTiles) {
      if (USE_WS) {
        __builtin_prefetch(wsHi + wsOffA(nt, 0, lane), 0, 1);
        __builtin_prefetch((lane < 16) ? (const void*)(wsLo + wsOffA(nt, 0, lane))
                                       : (const void*)(e2perm + wsOffE2(nt, 0, lane)), 0, 1);
      } else {
        __builtin_prefetch((lane < 16) ? (e1 + (size_t)(nt * 16 + c) * 64)
                                       : (e2 + (size_t)(nt * 16 + c) * 64), 0, 1);
      }
    }

    // A operands: E1 tile, 2 K-chunks, bf16 hi/lo (pre-split loads or inline).
    v16bf ah[2], al[2];
    if (USE_WS) {
      loadA(wsHi, wsLo, tile, 0, lane, ah[0], al[0]);
      loadA(wsHi, wsLo, tile, 1, lane, ah[1], al[1]);
    } else {
      buildA(e1, b0, 64, 0,  lane, ah[0], al[0]);
      buildA(e1, b0, 64, 32, lane, ah[1], al[1]);
    }

    // T = E1 @ W[k] : 4 n-tiles x 2 k-chunks x 3 split products = 24 WMMAs.
    v8f acc[4];
#pragma unroll
    for (int t = 0; t < 4; ++t) {
      v8f a = {};
      a = wmma3(ah[0], al[0], wh[t][0], wl[t][0], a);
      a = wmma3(ah[1], al[1], wh[t][1], wl[t][1], a);
      acc[t] = a;
    }

    // Row-wise dot with E2 in full fp32 on the native C/D layout.
    v8f p = {};
#pragma unroll
    for (int t = 0; t < 4; ++t) {
      if (USE_WS) {
        const float4* ep = (const float4*)(e2perm + wsOffE2(tile, t, lane));
        const float4 q0 = ep[0], q1 = ep[1];
        p[0] += acc[t][0] * q0.x;  p[1] += acc[t][1] * q0.y;
        p[2] += acc[t][2] * q0.z;  p[3] += acc[t][3] * q0.w;
        p[4] += acc[t][4] * q1.x;  p[5] += acc[t][5] * q1.y;
        p[6] += acc[t][6] * q1.z;  p[7] += acc[t][7] * q1.w;
      } else {
        const float* e2p = e2 + (size_t)(b0 + hoff) * 64 + t * 16 + c;
#pragma unroll
        for (int v = 0; v < 8; ++v)
          p[v] += acc[t][v] * e2p[(size_t)v * 64];
      }
    }

    // Wave-private LDS transpose: [row = v+hoff][class = c] partials,
    // lane c then sums its row's 16 classes -> bil[k, b0+c].
#pragma unroll
    for (int v = 0; v < 8; ++v)
      xpose[wv][hoff + v][c] = p[v];
    __asm__ volatile("s_wait_dscnt 0" ::: "memory");  // wave-internal LDS exchange

    if (lane < 16) {
      const float4* row = (const float4*)&xpose[wv][lane][0];  // 64-B aligned
      const float4 r0 = row[0], r1 = row[1], r2 = row[2], r3 = row[3];
      const float s = ((r0.x + r0.y) + (r0.z + r0.w)) + ((r1.x + r1.y) + (r1.z + r1.w))
                    + ((r2.x + r2.y) + (r2.z + r2.w)) + ((r3.x + r3.y) + (r3.z + r3.w));
      // Reference reshape: bil[k,b] -> out[k*(B/16) + b/16, b%16].
      const size_t R   = (size_t)k * nTiles + tile;
      const size_t idx = R * 16 + lane;
      const float ff = out[idx];                 // written by kernel 1
      out[idx] = tanhf(s + bs + ff);
    }
    __asm__ volatile("s_wait_dscnt 0" ::: "memory");  // reads done before reuse
  }
}

// ---------------------------------------------------------------------------
extern "C" void kernel_launch(void* const* d_in, const int* in_sizes, int n_in,
                              void* d_out, int out_size, void* d_ws, size_t ws_size,
                              hipStream_t stream) {
  const float* e1   = (const float*)d_in[0];   // [B, 64]
  const float* e2   = (const float*)d_in[1];   // [B, 64]
  const float* W    = (const float*)d_in[2];   // [16, 64, 64]
  const float* V    = (const float*)d_in[3];   // [128, 16]
  const float* bias = (const float*)d_in[4];   // [1, 64]
  float* out = (float*)d_out;                  // [B, 16]

  const int Bn = in_sizes[0] / 64;
  const int nTiles = Bn / 16;                  // 16384 for the reference shapes

  // Workspace: e1-hi (2 KB/tile) + e1-lo (2 KB/tile) + e2-perm (4 KB/tile).
  const size_t perA = (size_t)nTiles * 2048;
  const bool useWs = (d_ws != nullptr) && (ws_size >= 4 * perA);
  __bf16* wsHi   = (__bf16*)d_ws;
  __bf16* wsLo   = wsHi + perA / sizeof(__bf16);
  float*  e2perm = (float*)((char*)d_ws + 2 * perA);

  int g1 = (nTiles + 3) / 4;  if (g1 > 1024) g1 = 1024;  if (g1 < 1) g1 = 1;
  int g2 = nTiles / 4;        if (g2 > 4096) g2 = 4096;  if (g2 < 4) g2 = 4;
  g2 &= ~3;                                    // 4 waves/block -> multiple of 16 waves

  if (useWs) {
    int g0 = (nTiles + 3) / 4;  if (g0 > 1024) g0 = 1024;  if (g0 < 1) g0 = 1;
    ntn_prep_kernel<<<g0, 128, 0, stream>>>(e1, e2, wsHi, wsLo, e2perm, nTiles);
    ntn_ff_kernel<true><<<g1, 128, 0, stream>>>(e1, e2, V, out, wsHi, wsLo, nTiles);
    ntn_bil_kernel<true><<<g2, 128, 0, stream>>>(e1, e2, W, bias, out, wsHi, wsLo, e2perm, nTiles);
  } else {
    ntn_ff_kernel<false><<<g1, 128, 0, stream>>>(e1, e2, V, out, nullptr, nullptr, nTiles);
    ntn_bil_kernel<false><<<g2, 128, 0, stream>>>(e1, e2, W, bias, out, nullptr, nullptr,
                                                  nullptr, nTiles);
  }
}
